// CustomRNN_28321014350557
// MI455X (gfx1250) — matrix operand order repaired and checked by reading, not compiled, not run on hardware
//
#include <hip/hip_runtime.h>
#include <hip/hip_bf16.h>
#include <math.h>

// ---------------------------------------------------------------------------
// 2-layer tanh RNN on gfx1250 (MI455X).
//   h0' = tanh(x_t @ Wx0 + h0 @ Wh0 + b0)   (B=64,V=128,H=1024)
//   h1' = tanh(h0' @ Wx1 + h1 @ Wh1 + b1)
//
// 4 independent batch groups (16 rows = one WMMA M-tile) x 16 WGs per group
// (64-col H slice each) = 64 persistent WGs. Sync confined to 16-WG groups
// (cluster barrier fast path + phase-counter atomic barrier).
// Matmuls: V_WMMA_F32_16X16X32_BF16, two interleaved accumulators.
// Per-WG weight slices Wh0/Wh1/Wx0 (272KB bf16) are DMA'd into LDS ONCE via
// the Tensor Data Mover (tensor_load_to_lds, TENSORcnt) and served from LDS
// every step; only the Wx1 slice streams from L2.
// ---------------------------------------------------------------------------

typedef __bf16 bf16_t;
typedef __attribute__((ext_vector_type(16))) __bf16 v16bf;
typedef __attribute__((ext_vector_type(8)))  __bf16 v8bf;
typedef __attribute__((ext_vector_type(8)))  float  v8f;
typedef __attribute__((ext_vector_type(4)))  unsigned int u32x4;
typedef __attribute__((ext_vector_type(8)))  int i32x8;
typedef __attribute__((ext_vector_type(4)))  int i32x4;

#define B_    64
#define T_    512
#define V_    128
#define H_    1024
#define NG_   4      // batch groups (16 rows each)
#define WPG_  16     // workgroups per group (64 cols each)
#define NT_   64     // total 16-col N tiles in H
#define KTV_  4      // K tiles of 32 covering V=128
#define KTH_  32     // K tiles of 32 covering H=1024

// LDS layout (bf16 elements): [Wh0: 32*4*512][Wh1: 32*4*512][Wx0: 4*4*512]
#define LDS_WH0_ELEM  0
#define LDS_WH1_ELEM  65536
#define LDS_WX0_ELEM  131072
#define LDS_TOT_ELEM  139264            // 278528 bytes (<320KB WGP LDS)

#if defined(__has_builtin)
#  if __has_builtin(__builtin_amdgcn_tensor_load_to_lds)
#    define HAVE_TDM 1
#  endif
#endif

// ------------------------- prep kernels ------------------------------------

// fp32 [K][1024] weight -> bf16 B-fragment layout (ISA 7.12.2, 16-bit B):
// dst[((kt*64+nt)*32+lane)*16+slot]; K=kt*32+(lane>>4)*8+(slot<8?slot:slot+8);
// n = nt*16 + (lane&15).
__global__ void swz_w_kernel(const float* __restrict__ src,
                             bf16_t* __restrict__ dst, int total) {
  int i = blockIdx.x * 256 + threadIdx.x;
  if (i >= total) return;
  int slot = i & 15;
  int lane = (i >> 4) & 31;
  int nt   = (i >> 9) & 63;
  int kt   = i >> 15;
  int n = nt * 16 + (lane & 15);
  int k = kt * 32 + ((lane >> 4) << 3) + (slot < 8 ? slot : slot + 8);
  dst[i] = (bf16_t)src[(size_t)k * H_ + n];
}

// x (B,T,V) fp32 -> per-(t, batch-group) bf16 A-fragments (16x32 layout).
__global__ void swz_x_kernel(const float* __restrict__ x,
                             bf16_t* __restrict__ dst) {
  int i = blockIdx.x * 256 + threadIdx.x;
  if (i >= T_ * NG_ * KTV_ * 512) return;
  int slot = i & 15;
  int lane = (i >> 4) & 31;
  int kt   = (i >> 9) & 3;
  int g    = (i >> 11) & 3;
  int t    = i >> 13;
  int b    = g * 16 + (lane & 15);
  int half = lane >> 4;
  int koff = half * 8 + (slot < 8 ? slot : slot + 8);
  int k    = kt * 32 + koff;
  dst[i] = (bf16_t)x[(size_t)b * T_ * V_ + (size_t)t * V_ + k];
}

__global__ void init_state_kernel(const float* __restrict__ h_in,
                                  bf16_t* __restrict__ H0,
                                  bf16_t* __restrict__ H1) {
  int i = blockIdx.x * 256 + threadIdx.x;
  if (i >= B_ * H_) return;
  H0[i] = (bf16_t)h_in[i];
  H1[i] = (bf16_t)h_in[B_ * H_ + i];
}

__global__ void copy_c_kernel(const float* __restrict__ c,
                              float* __restrict__ out_c) {
  int i = blockIdx.x * 256 + threadIdx.x;
  if (i < 2 * B_ * H_) out_c[i] = c[i];
}

__global__ void zero_bar_kernel(unsigned* __restrict__ bar) {
  if (threadIdx.x < NG_) bar[threadIdx.x] = 0u;
}

// ------------------------- fragment loaders --------------------------------

__device__ __forceinline__ v16bf load_bfrag(const bf16_t* __restrict__ W,
                                            int kt, int ntile, int lane) {
  return *(const v16bf*)(W + ((((size_t)kt * NT_) + ntile) * 32 + lane) * 16);
}

// LDS slice layout: [kt][lt(0..3)][lane][16]
__device__ __forceinline__ v16bf load_bfrag_lds(const bf16_t* Lb, int kt,
                                                int lt, int lane) {
  return *(const v16bf*)(Lb + (((size_t)kt * 4 + lt) * 32 + lane) * 16);
}

__device__ __forceinline__ v16bf load_afrag_state(const bf16_t* __restrict__ S,
                                                  int row0, int ln, int half,
                                                  int k0) {
  const bf16_t* p = S + (size_t)(row0 + ln) * H_ + k0 + half * 8;
  v8bf lo = *(const v8bf*)p;         // slots 0-7
  v8bf hi = *(const v8bf*)(p + 16);  // slots 8-15
  return __builtin_shufflevector(lo, hi, 0, 1, 2, 3, 4, 5, 6, 7,
                                 8, 9, 10, 11, 12, 13, 14, 15);
}

__device__ __forceinline__ v8f wmma_bf16(v16bf a, v16bf b, v8f c) {
  return __builtin_amdgcn_wmma_f32_16x16x32_bf16(
      false, a, false, b, (short)0, c, false, false);
}

// Phase-counting group barrier + CDNA5 cluster-barrier fast path.
__device__ __forceinline__ void group_barrier(unsigned* __restrict__ ctr,
                                              unsigned target) {
  __threadfence();
  __syncthreads();
  __builtin_amdgcn_s_cluster_barrier();
  if (threadIdx.x == 0) {
    __hip_atomic_fetch_add(ctr, 1u, __ATOMIC_RELEASE, __HIP_MEMORY_SCOPE_AGENT);
    while (__hip_atomic_load(ctr, __ATOMIC_ACQUIRE, __HIP_MEMORY_SCOPE_AGENT) <
           target) {
      __builtin_amdgcn_s_sleep(1);
    }
  }
  __syncthreads();
  __threadfence();
}

// ------------------------- TDM descriptor ----------------------------------
// 2D tile load: `rows` x 4KB, global row stride 64KB, data_size = 4B.
// D# per CDNA5 ISA sect. 8.3/8.4 (count=1, type=2).
__device__ __forceinline__ void tdm_load_2d(unsigned lds_byte_off,
                                            const void* src, unsigned rows) {
#if HAVE_TDM
  unsigned long long ga = (unsigned long long)(uintptr_t)src;
  u32x4 g0;
  g0.x = 1u;                                        // count=1, user mode
  g0.y = lds_byte_off;                              // lds_addr
  g0.z = (unsigned)(ga & 0xFFFFFFFFu);              // global_addr[31:0]
  g0.w = (unsigned)((ga >> 32) & 0x01FFFFFFu) | (2u << 30);  // addr[56:32]|type=2
  i32x8 g1;
  g1[0] = (2 << 16);          // workgroup_mask=0, data_size=4B, no flags
  g1[1] = (1024 << 16);       // atomic_bar_addr=0 | tensor_dim0.lo = 1024
  g1[2] = (int)(rows << 16);  // tensor_dim0.hi = 0 | tensor_dim1.lo = rows
  g1[3] = (1024 << 16);       // tensor_dim1.hi = 0 | tile_dim0 = 1024 units
  g1[4] = (int)rows;          // tile_dim1 = rows, tile_dim2 = 0
  g1[5] = 16384;              // tensor_dim0_stride = 16384 units (64KB)
  g1[6] = 0;                  // stride.hi | tensor_dim1_stride.lo (unused)
  g1[7] = 0;
  i32x4 z4 = {0, 0, 0, 0};
#if __clang_major__ >= 23
  i32x8 z8 = {0, 0, 0, 0, 0, 0, 0, 0};
  __builtin_amdgcn_tensor_load_to_lds(g0, g1, z4, z4, z8, 0);
#else
  __builtin_amdgcn_tensor_load_to_lds(g0, g1, z4, z4, 0);
#endif
#endif
}

// ------------------------- persistent RNN kernel ---------------------------

__global__ __launch_bounds__(128, 1) void rnn_step_kernel(
    const bf16_t* __restrict__ Wx0s, const bf16_t* __restrict__ Wh0s,
    const bf16_t* __restrict__ Wx1s, const bf16_t* __restrict__ Wh1s,
    const bf16_t* __restrict__ Xs,
    const float* __restrict__ b0v, const float* __restrict__ b1v,
    bf16_t* __restrict__ H0, bf16_t* __restrict__ H1,
    float* __restrict__ outs, float* __restrict__ hlast,
    unsigned* __restrict__ bar) {
  extern __shared__ bf16_t smem[];   // 278528 B dynamic LDS

  const int g     = blockIdx.x >> 4;   // batch group 0..3
  const int w     = blockIdx.x & 15;   // WG-in-group -> 64-col slice
  const int wave  = threadIdx.x >> 5;  // 0..3 (local n-tile)
  const int lane  = threadIdx.x & 31;
  const int ln    = lane & 15;
  const int half  = lane >> 4;

  const int row0    = g * 16;
  const int ntile   = w * 4 + wave;    // global 16-col tile
  const int col     = ntile * 16 + ln;
  const int rowbase = half * 8;

  const bf16_t* ldsWh0 = smem + LDS_WH0_ELEM;
  const bf16_t* ldsWh1 = smem + LDS_WH1_ELEM;
  const bf16_t* ldsWx0 = smem + LDS_WX0_ELEM;

  // ---- one-time weight-slice residency in LDS ----
#if HAVE_TDM
  if (wave == 0) {   // TDM is a DMA; one descriptor per matrix slice
    tdm_load_2d(LDS_WH0_ELEM * 2, Wh0s + (size_t)(w * 4) * 512, KTH_);
    tdm_load_2d(LDS_WH1_ELEM * 2, Wh1s + (size_t)(w * 4) * 512, KTH_);
    tdm_load_2d(LDS_WX0_ELEM * 2, Wx0s + (size_t)(w * 4) * 512, KTV_);
    __builtin_amdgcn_s_wait_tensorcnt(0);
  }
#else
  // Fallback: cooperative copy (gl loads + ds stores).
  for (int e = threadIdx.x * 8; e < KTH_ * 4 * 512; e += 128 * 8) {
    int kt = e >> 11, rem = e & 2047;
    *(v8bf*)(smem + LDS_WH0_ELEM + e) =
        *(const v8bf*)(Wh0s + ((size_t)kt * NT_ + w * 4) * 512 + rem);
    *(v8bf*)(smem + LDS_WH1_ELEM + e) =
        *(const v8bf*)(Wh1s + ((size_t)kt * NT_ + w * 4) * 512 + rem);
  }
  for (int e = threadIdx.x * 8; e < KTV_ * 4 * 512; e += 128 * 8) {
    int kt = e >> 11, rem = e & 2047;
    *(v8bf*)(smem + LDS_WX0_ELEM + e) =
        *(const v8bf*)(Wx0s + ((size_t)kt * NT_ + w * 4) * 512 + rem);
  }
#endif
  __syncthreads();

  unsigned* ctr = bar + g;
  const float bias0 = b0v[col];
  const float bias1 = b1v[col];
  unsigned phase = 0;

  for (int t = 0; t < T_; ++t) {
    const int p = t & 1;
    const bf16_t* H0r = H0 + (size_t)p * B_ * H_;
    bf16_t*       H0w = H0 + (size_t)(p ^ 1) * B_ * H_;
    const bf16_t* H1r = H1 + (size_t)p * B_ * H_;
    bf16_t*       H1w = H1 + (size_t)(p ^ 1) * B_ * H_;

    // ------------------ layer 0: x_t@Wx0 + h0@Wh0 ------------------
    v8f acc0 = {0.f, 0.f, 0.f, 0.f, 0.f, 0.f, 0.f, 0.f};
    v8f acc1 = {0.f, 0.f, 0.f, 0.f, 0.f, 0.f, 0.f, 0.f};
    const bf16_t* xt = Xs + (((size_t)t * NG_ + g) * KTV_) * 512;
#pragma unroll
    for (int kt = 0; kt < KTV_; ++kt) {
      v16bf a  = *(const v16bf*)(xt + (size_t)kt * 512 + lane * 16);
      v16bf bb = load_bfrag_lds(ldsWx0, kt, wave, lane);
      if (kt & 1) acc1 = wmma_bf16(a, bb, acc1);
      else        acc0 = wmma_bf16(a, bb, acc0);
    }
#pragma unroll 4
    for (int kt = 0; kt < KTH_; ++kt) {
      v16bf a  = load_afrag_state(H0r, row0, ln, half, kt * 32);
      v16bf bb = load_bfrag_lds(ldsWh0, kt, wave, lane);
      if (kt & 1) acc1 = wmma_bf16(a, bb, acc1);
      else        acc0 = wmma_bf16(a, bb, acc0);
    }
    {
      v8f acc = acc0 + acc1;
      float h0v[8];
#pragma unroll
      for (int r = 0; r < 8; ++r) {
        h0v[r] = tanhf(acc[r] + bias0);
        H0w[(size_t)(row0 + rowbase + r) * H_ + col] = (bf16_t)h0v[r];
      }
      if (t == T_ - 1) {
#pragma unroll
        for (int r = 0; r < 8; ++r)
          hlast[(size_t)(row0 + rowbase + r) * H_ + col] = h0v[r];
      }
    }

    ++phase;
    group_barrier(ctr, 16u * phase);   // h0' fully published in group

    // ------------------ layer 1: h0'@Wx1 + h1@Wh1 ------------------
    acc0 = (v8f){0.f, 0.f, 0.f, 0.f, 0.f, 0.f, 0.f, 0.f};
    acc1 = (v8f){0.f, 0.f, 0.f, 0.f, 0.f, 0.f, 0.f, 0.f};
#pragma unroll 4
    for (int kt = 0; kt < KTH_; ++kt) {
      v16bf a  = load_afrag_state(H0w, row0, ln, half, kt * 32);
      v16bf bb = load_bfrag(Wx1s, kt, ntile, lane);   // streamed from L2
      if (kt & 1) acc1 = wmma_bf16(a, bb, acc1);
      else        acc0 = wmma_bf16(a, bb, acc0);
    }
#pragma unroll 4
    for (int kt = 0; kt < KTH_; ++kt) {
      v16bf a  = load_afrag_state(H1r, row0, ln, half, kt * 32);
      v16bf bb = load_bfrag_lds(ldsWh1, kt, wave, lane);
      if (kt & 1) acc1 = wmma_bf16(a, bb, acc1);
      else        acc0 = wmma_bf16(a, bb, acc0);
    }
    {
      v8f acc = acc0 + acc1;
#pragma unroll
      for (int r = 0; r < 8; ++r) {
        const int b = row0 + rowbase + r;
        float v = tanhf(acc[r] + bias1);
        outs[(size_t)b * T_ * H_ + (size_t)t * H_ + col] = v;
        H1w[(size_t)b * H_ + col] = (bf16_t)v;
        if (t == T_ - 1) hlast[(size_t)B_ * H_ + (size_t)b * H_ + col] = v;
      }
    }

    ++phase;
    group_barrier(ctr, 16u * phase);   // step complete; ping-pong safe
  }
}

// ------------------------- host launch -------------------------------------

extern "C" void kernel_launch(void* const* d_in, const int* in_sizes, int n_in,
                              void* d_out, int out_size, void* d_ws,
                              size_t ws_size, hipStream_t stream) {
  (void)in_sizes; (void)n_in; (void)out_size; (void)ws_size;
  const float* x   = (const float*)d_in[0];
  const float* h   = (const float*)d_in[1];
  const float* c   = (const float*)d_in[2];
  const float* Wx0 = (const float*)d_in[3];
  const float* Wh0 = (const float*)d_in[4];
  const float* b0  = (const float*)d_in[5];
  const float* Wx1 = (const float*)d_in[6];
  const float* Wh1 = (const float*)d_in[7];
  const float* b1  = (const float*)d_in[8];

  bf16_t* ws = (bf16_t*)d_ws;
  size_t o = 0;
  bf16_t* Wx0s = ws + o; o += (size_t)KTV_ * NT_ * 512;
  bf16_t* Wh0s = ws + o; o += (size_t)KTH_ * NT_ * 512;
  bf16_t* Wx1s = ws + o; o += (size_t)KTH_ * NT_ * 512;
  bf16_t* Wh1s = ws + o; o += (size_t)KTH_ * NT_ * 512;
  bf16_t* Xs   = ws + o; o += (size_t)T_ * NG_ * KTV_ * 512;
  bf16_t* H0   = ws + o; o += 2ull * B_ * H_;
  bf16_t* H1   = ws + o; o += 2ull * B_ * H_;
  unsigned* bar = (unsigned*)(ws + o);

  float* outs  = (float*)d_out;
  float* hlast = outs + (size_t)B_ * T_ * H_;
  float* outc  = hlast + 2ull * B_ * H_;

  zero_bar_kernel<<<1, 32, 0, stream>>>(bar);

  int nWx0 = KTV_ * NT_ * 512;
  int nWh  = KTH_ * NT_ * 512;
  swz_w_kernel<<<(nWx0 + 255) / 256, 256, 0, stream>>>(Wx0, Wx0s, nWx0);
  swz_w_kernel<<<(nWh  + 255) / 256, 256, 0, stream>>>(Wh0, Wh0s, nWh);
  swz_w_kernel<<<(nWh  + 255) / 256, 256, 0, stream>>>(Wx1, Wx1s, nWh);
  swz_w_kernel<<<(nWh  + 255) / 256, 256, 0, stream>>>(Wh1, Wh1s, nWh);

  int nX = T_ * NG_ * KTV_ * 512;
  swz_x_kernel<<<(nX + 255) / 256, 256, 0, stream>>>(x, Xs);

  init_state_kernel<<<(B_ * H_ + 255) / 256, 256, 0, stream>>>(h, H0, H1);
  copy_c_kernel<<<(2 * B_ * H_ + 255) / 256, 256, 0, stream>>>(c, outc);

  size_t ldsBytes = (size_t)LDS_TOT_ELEM * sizeof(bf16_t);  // 278528
  (void)hipFuncSetAttribute((const void*)rnn_step_kernel,
                            hipFuncAttributeMaxDynamicSharedMemorySize,
                            (int)ldsBytes);
  rnn_step_kernel<<<NG_ * WPG_, 128, ldsBytes, stream>>>(
      Wx0s, Wh0s, Wx1s, Wh1s, Xs, b0, b1, H0, H1, outs, hlast, bar);
}
